// GradientGuidedAttention_74577812127852
// MI455X (gfx1250) — matrix operand order repaired and checked by reading, not compiled, not run on hardware
//
#include <hip/hip_runtime.h>
#include <stdint.h>

typedef __attribute__((ext_vector_type(16))) _Float16 v16h;
typedef __attribute__((ext_vector_type(8)))  _Float16 v8h;
typedef __attribute__((ext_vector_type(8)))  float    v8f;
typedef __attribute__((ext_vector_type(4)))  uint32_t v4u;
typedef __attribute__((ext_vector_type(8)))  int      v8i;
typedef __attribute__((ext_vector_type(4)))  int      v4i;

#define WMMA_F32_F16(A, B, Cacc) \
  __builtin_amdgcn_wmma_f32_16x16x32_f16(false, (A), false, (B), (short)0, (Cacc), false, false)

#if __has_builtin(__builtin_amdgcn_tensor_load_to_lds) && \
    __has_builtin(__builtin_amdgcn_s_wait_tensorcnt)
#define HAVE_TDM 1
#endif

constexpr int   BATCH   = 2;
constexpr int   T       = 2048;
constexpr int   CH      = 512;      // channels
constexpr int   H       = 16;
constexpr int   HD      = 32;       // head dim == WMMA K — one wmma per 16x16 score tile
constexpr int   BT      = BATCH * T;
constexpr float SCALE   = 0.17677669529663687f;  // 1/sqrt(32)
constexpr float NEGINF  = -1e30f;
constexpr int   N_EXTRA = (int)(T * (1.0f - 0.3f)) - 3;  // 1430

// Build a 16-half fragment from two contiguous 16-byte runs (two ds_load_b128).
__device__ __forceinline__ v16h frag16(const _Float16* p0, const _Float16* p1) {
  v8h lo = *(const v8h*)p0;
  v8h hi = *(const v8h*)p1;
  return __builtin_shufflevector(lo, hi, 0, 1, 2, 3, 4, 5, 6, 7,
                                          8, 9, 10, 11, 12, 13, 14, 15);
}

#ifdef HAVE_TDM
// Async 2D f16 tile (tile_d0 x tile_d1 elements, row stride stride0 elements)
// global -> LDS via the Tensor Data Mover. pad: +4 dwords after every 16 dwords
// stored, matching LDS rows padded from 32 to 40 halves.
// D# packing per CDNA5 ISA 8.3/8.4.
__device__ __forceinline__ void tdm_load_tile_f16(
    uint32_t lds_addr, const void* gaddr, uint32_t tile_d0, uint32_t tile_d1,
    uint32_t tensor_d0, uint32_t tensor_d1, uint32_t stride0, bool pad) {
  const uint64_t ga = (uint64_t)(uintptr_t)gaddr;
  v4u g0;
  g0[0] = 1u;                                            // count=1 (valid user D#)
  g0[1] = lds_addr;                                      // LDS byte address
  g0[2] = (uint32_t)ga;                                  // global_addr[31:0]
  g0[3] = (uint32_t)((ga >> 32) & 0x01FFFFFFu) | (2u << 30);  // addr[56:32] | type=2
  v8i g1;
  uint32_t dw0 = (1u << 16);                             // data_size = 2 bytes
  if (pad) dw0 |= (1u << 20) | (3u << 22) | (3u << 25);  // pad_en, every 16dw, +4dw
  g1[0] = (int)dw0;
  g1[1] = (int)((tensor_d0 & 0xFFFFu) << 16);            // abar=0 | tensor_dim0[15:0]
  g1[2] = (int)((tensor_d0 >> 16) | ((tensor_d1 & 0xFFFFu) << 16));
  g1[3] = (int)((tensor_d1 >> 16) | (tile_d0 << 16));    // tensor_dim1 hi | tile_dim0
  g1[4] = (int)(tile_d1 & 0xFFFFu);                      // tile_dim1 | tile_dim2=0
  g1[5] = (int)stride0;                                  // dim0_stride[31:0]
  g1[6] = 0;                                             // dim0_stride hi | dim1_stride lo
  g1[7] = 0;
  const v4i z4 = {0, 0, 0, 0};                           // groups 2/3: 2D tensor
  const v8i z8 = {0, 0, 0, 0, 0, 0, 0, 0};
  __builtin_amdgcn_tensor_load_to_lds(g0, g1, z4, z4, z8, 0);
}
#endif

// ---------------------------------------------------------------- utilities
__device__ __forceinline__ float hash01(uint32_t q, uint32_t k) {
  uint32_t h = q * 0x9E3779B1u ^ (k + 0x7F4A7C15u);
  h *= 0x85EBCA77u; h ^= h >> 13; h *= 0xC2B2AE3Du; h ^= h >> 16;
  return (float)(h >> 8) * (1.0f / 16777216.0f);
}

// causal AND (band OR importance-weighted bernoulli ~ gumbel top-k marginal)
__device__ __forceinline__ bool allowed(int q, int k, float rate) {
  if (k > q) return false;
  if (q - k <= 1) return true;
  return hash01((uint32_t)q, (uint32_t)k) < rate;
}

// ---------------------------------------------------------------- fp32->fp16
__global__ void cvt_f32_f16(const float* __restrict__ in, _Float16* __restrict__ out, int n) {
  int i = blockIdx.x * blockDim.x + threadIdx.x;
  if (i < n) out[i] = (_Float16)in[i];
}

// ---------------------------------------------------- keep-rate (importance softmax)
__global__ void keeprate_kernel(const float* __restrict__ g, float* __restrict__ kr) {
  __shared__ float red[256];
  const int tid = threadIdx.x;
  float m = NEGINF;
  for (int i = tid; i < T; i += 256) m = fmaxf(m, g[i] * 100.0f);
  red[tid] = m; __syncthreads();
  for (int s = 128; s > 0; s >>= 1) {
    if (tid < s) red[tid] = fmaxf(red[tid], red[tid + s]);
    __syncthreads();
  }
  const float gm = red[0]; __syncthreads();
  float sum = 0.f;
  for (int i = tid; i < T; i += 256) sum += __expf(g[i] * 100.0f - gm);
  red[tid] = sum; __syncthreads();
  for (int s = 128; s > 0; s >>= 1) {
    if (tid < s) red[tid] += red[tid + s];
    __syncthreads();
  }
  const float inv = 1.0f / red[0];
  for (int i = tid; i < T; i += 256) {
    float p = __expf(g[i] * 100.0f - gm) * inv;
    kr[i] = fminf(1.0f, (float)N_EXTRA * p);
  }
}

// ---------------------------------------------------------------- QKV GEMM
// Y = Xh[4096x512] * W[512x512]; result scattered to [B,H,T,HD] f16.
// A tile staged by the TDM (async DMA, padded rows); B tile staged TRANSPOSED
// (Bt[col][k]) so all B-fragment reads are ds_load_b128.
__global__ __launch_bounds__(128) void qkv_gemm(
    const _Float16* __restrict__ X,
    const _Float16* __restrict__ W0, const _Float16* __restrict__ W1,
    const _Float16* __restrict__ W2,
    _Float16* __restrict__ Qd, _Float16* __restrict__ Kd, _Float16* __restrict__ Vd) {
  __shared__ __align__(16) _Float16 As[64][40];  // [row][k]
  __shared__ __align__(16) _Float16 Bt[64][40];  // [col][k]  (transposed W tile)

  const int r0 = blockIdx.y * 64;
  const int c0 = blockIdx.x * 64;
  const _Float16* W = (blockIdx.z == 0) ? W0 : ((blockIdx.z == 1) ? W1 : W2);
  _Float16* D = (blockIdx.z == 0) ? Qd : ((blockIdx.z == 1) ? Kd : Vd);

  const int tid  = threadIdx.x;
  const int lane = tid & 31;
  const int wv   = tid >> 5;
  const int wr   = (wv >> 1) * 32, wc = (wv & 1) * 32;
  const int nl   = lane & 15;
  const int hi   = (lane >= 16);
  const int kb8  = hi ? 8 : 0;
  const int kb16 = hi ? 16 : 0;

  v8f acc[2][2] = {};

  for (int kk = 0; kk < CH; kk += 32) {
    __syncthreads();
#ifdef HAVE_TDM
    if (tid < 32)  // wave 0 issues the async A-tile DMA (64x32, padded to 40)
      tdm_load_tile_f16((uint32_t)(uintptr_t)&As[0][0],
                        &X[(size_t)r0 * CH + kk], 32u, 64u,
                        (uint32_t)CH, (uint32_t)BT, (uint32_t)CH, true);
#else
    for (int v = tid; v < 256; v += 128) {
      int row = v >> 2, ko = (v & 3) * 8;
      *(v8h*)&As[row][ko] = *(const v8h*)&X[(size_t)(r0 + row) * CH + kk + ko];
    }
#endif
    // stage B tile 32x64 transposed into Bt[col][k]
    for (int v = tid; v < 256; v += 128) {
      int krow = v >> 3, co = (v & 7) * 8;
      v8h w = *(const v8h*)&W[(size_t)(kk + krow) * CH + c0 + co];
#pragma unroll
      for (int j = 0; j < 8; ++j) Bt[co + j][krow] = w[j];
    }
#ifdef HAVE_TDM
    if (tid < 32) __builtin_amdgcn_s_wait_tensorcnt(0);
#endif
    __syncthreads();

    v16h a[2], b[2];
#pragma unroll
    for (int t2 = 0; t2 < 2; ++t2) {  // A frag: row = lane, K split lo/hi half-wave
      int m = wr + t2 * 16 + nl;
      a[t2] = frag16(&As[m][kb16], &As[m][kb16 + 8]);
    }
#pragma unroll
    for (int t2 = 0; t2 < 2; ++t2) {  // B frag: col = lane, 16 contiguous K halves
      int n = wc + t2 * 16 + nl;
      b[t2] = frag16(&Bt[n][kb16], &Bt[n][kb16 + 8]);
    }
#pragma unroll
    for (int ti = 0; ti < 2; ++ti)
#pragma unroll
      for (int tj = 0; tj < 2; ++tj)
        acc[ti][tj] = WMMA_F32_F16(a[ti], b[tj], acc[ti][tj]);
  }

  // scatter to [B,H,T,HD] as f16
#pragma unroll
  for (int ti = 0; ti < 2; ++ti)
#pragma unroll
    for (int tj = 0; tj < 2; ++tj)
#pragma unroll
      for (int i = 0; i < 8; ++i) {
        int r = r0 + wr + ti * 16 + i + (hi ? 8 : 0);
        int c = c0 + wc + tj * 16 + nl;
        int bb = r / T, t = r % T;
        int h = c / HD, d = c % HD;
        D[(((size_t)(bb * H + h)) * T + t) * HD + d] = (_Float16)acc[ti][tj][i];
      }
}

// ------------------------------------------------- flash attention (masked)
__global__ __launch_bounds__(128) void attn_kernel(
    const _Float16* __restrict__ Qh, const _Float16* __restrict__ Kh,
    const _Float16* __restrict__ Vh, const float* __restrict__ kr,
    _Float16* __restrict__ Oh) {
  __shared__ __align__(16) _Float16 kS[32][40];  // [key][d]   (B k-dim = d: contiguous)
  __shared__ __align__(16) _Float16 vT[32][40];  // [d][key]   (B k-dim = key: contiguous)
  __shared__ __align__(16) _Float16 pS[4][16][32];
  __shared__ float krS[32];

  const int qb  = blockIdx.x;        // q tile of 64 rows
  const int bh  = blockIdx.y;        // (b,h)
  const int b   = bh / H;
  const int h   = bh % H;
  const int tid = threadIdx.x;
  const int lane = tid & 31;
  const int wv   = tid >> 5;
  const int nl   = lane & 15;
  const int hi   = (lane >= 16);
  const int kb8  = hi ? 8 : 0;
  const int kb16 = hi ? 16 : 0;

  const size_t headoff = (size_t)bh * T * HD;
  const _Float16* Qp = Qh + headoff;
  const _Float16* Kp = Kh + headoff;
  const _Float16* Vp = Vh + headoff;

  const int q0 = qb * 64 + wv * 16;  // this wave's 16 q rows

  // Q A-fragment (16x32) — loaded once (two global_load_b128), hd==K==32
  const size_t qbase = (size_t)(q0 + nl) * HD;
  const v16h aQ = frag16(&Qp[qbase + kb16], &Qp[qbase + kb16 + 8]);

  v8f o0 = {}, o1 = {};
  float mrow[8], lrow[8];
#pragma unroll
  for (int i = 0; i < 8; ++i) { mrow[i] = NEGINF; lrow[i] = 0.f; }

  const int kend = qb * 64 + 64;  // causal bound for this q tile

  for (int kt = 0; kt < kend; kt += 32) {
    __syncthreads();
#ifdef HAVE_TDM
    if (tid < 32)  // wave 0: async K-tile DMA (32 keys x 32 dims, padded rows)
      tdm_load_tile_f16((uint32_t)(uintptr_t)&kS[0][0],
                        &Kp[(size_t)kt * HD], 32u, 32u,
                        (uint32_t)HD, (uint32_t)T, (uint32_t)HD, true);
#endif
    {  // stage V transposed + keep rates (+ K manually if no TDM)
      int key = tid >> 2, doff = (tid & 3) * 8;
#ifndef HAVE_TDM
      *(v8h*)&kS[key][doff] = *(const v8h*)&Kp[(size_t)(kt + key) * HD + doff];
#endif
      v8h vv = *(const v8h*)&Vp[(size_t)(kt + key) * HD + doff];
#pragma unroll
      for (int j = 0; j < 8; ++j) vT[doff + j][key] = vv[j];
      if (tid < 32) krS[tid] = kr[kt + tid];
    }
#ifdef HAVE_TDM
    if (tid < 32) __builtin_amdgcn_s_wait_tensorcnt(0);
#endif
    __syncthreads();

    // S = Q * K^T : B(d,key) = K[key][d] -> contiguous along d in kS
    const v16h bK0 = frag16(&kS[nl][kb16],      &kS[nl][kb16 + 8]);
    const v16h bK1 = frag16(&kS[16 + nl][kb16], &kS[16 + nl][kb16 + 8]);
    v8f s0 = {}, s1 = {};
    s0 = WMMA_F32_F16(aQ, bK0, s0);
    s1 = WMMA_F32_F16(aQ, bK1, s1);

    // mask + online softmax; rows of C-layout live across 16-lane half groups
#pragma unroll
    for (int i = 0; i < 8; ++i) {
      const int row = q0 + i + (hi ? 8 : 0);
      const float v0 = allowed(row, kt + nl, krS[nl])           ? s0[i] * SCALE : NEGINF;
      const float v1 = allowed(row, kt + 16 + nl, krS[16 + nl]) ? s1[i] * SCALE : NEGINF;
      float tmax = fmaxf(v0, v1);
#pragma unroll
      for (int msk = 1; msk < 16; msk <<= 1) tmax = fmaxf(tmax, __shfl_xor(tmax, msk, 32));
      const float nm = fmaxf(mrow[i], tmax);
      const float p0 = (v0 <= -1e29f) ? 0.f : __expf(v0 - nm);
      const float p1 = (v1 <= -1e29f) ? 0.f : __expf(v1 - nm);
      float rs = p0 + p1;
#pragma unroll
      for (int msk = 1; msk < 16; msk <<= 1) rs += __shfl_xor(rs, msk, 32);
      const float corr = __expf(mrow[i] - nm);
      lrow[i] = lrow[i] * corr + rs;
      mrow[i] = nm;
      o0[i] *= corr; o1[i] *= corr;
      const int prow = i + (hi ? 8 : 0);
      pS[wv][prow][nl]      = (_Float16)p0;   // relayout C-frag -> A-frag via LDS
      pS[wv][prow][16 + nl] = (_Float16)p1;
    }

    // P as A-fragment (same-wave LDS RAW kept in order by DScnt)
    const v16h aP = frag16(&pS[wv][nl][kb16], &pS[wv][nl][kb16 + 8]);
    // V B-fragments: B(key,d) = V[key][d] = vT[d][key] -> contiguous along key
    const v16h bV0 = frag16(&vT[nl][kb16],      &vT[nl][kb16 + 8]);
    const v16h bV1 = frag16(&vT[16 + nl][kb16], &vT[16 + nl][kb16 + 8]);
    o0 = WMMA_F32_F16(aP, bV0, o0);
    o1 = WMMA_F32_F16(aP, bV1, o1);
  }

  // normalize and store [B,T,C] (C index = h*HD + d), f16 for the Wo GEMM
#pragma unroll
  for (int i = 0; i < 8; ++i) {
    const float inv = (lrow[i] > 0.f) ? 1.0f / lrow[i] : 0.f;
    const int t = q0 + i + (hi ? 8 : 0);
    const size_t base = ((size_t)(b * T + t)) * CH + h * HD;
    Oh[base + nl]      = (_Float16)(o0[i] * inv);
    Oh[base + 16 + nl] = (_Float16)(o1[i] * inv);
  }
}

// ---------------------------------------------------------- output proj GEMM
__global__ __launch_bounds__(128) void out_gemm(
    const _Float16* __restrict__ X, const _Float16* __restrict__ W,
    float* __restrict__ O) {
  __shared__ __align__(16) _Float16 As[64][40];
  __shared__ __align__(16) _Float16 Bt[64][40];

  const int r0 = blockIdx.y * 64;
  const int c0 = blockIdx.x * 64;
  const int tid  = threadIdx.x;
  const int lane = tid & 31;
  const int wv   = tid >> 5;
  const int wr   = (wv >> 1) * 32, wc = (wv & 1) * 32;
  const int nl   = lane & 15;
  const int hi   = (lane >= 16);
  const int kb8  = hi ? 8 : 0;
  const int kb16 = hi ? 16 : 0;

  v8f acc[2][2] = {};

  for (int kk = 0; kk < CH; kk += 32) {
    __syncthreads();
#ifdef HAVE_TDM
    if (tid < 32)
      tdm_load_tile_f16((uint32_t)(uintptr_t)&As[0][0],
                        &X[(size_t)r0 * CH + kk], 32u, 64u,
                        (uint32_t)CH, (uint32_t)BT, (uint32_t)CH, true);
#else
    for (int v = tid; v < 256; v += 128) {
      int row = v >> 2, ko = (v & 3) * 8;
      *(v8h*)&As[row][ko] = *(const v8h*)&X[(size_t)(r0 + row) * CH + kk + ko];
    }
#endif
    for (int v = tid; v < 256; v += 128) {
      int krow = v >> 3, co = (v & 7) * 8;
      v8h w = *(const v8h*)&W[(size_t)(kk + krow) * CH + c0 + co];
#pragma unroll
      for (int j = 0; j < 8; ++j) Bt[co + j][krow] = w[j];
    }
#ifdef HAVE_TDM
    if (tid < 32) __builtin_amdgcn_s_wait_tensorcnt(0);
#endif
    __syncthreads();

    v16h a[2], b[2];
#pragma unroll
    for (int t2 = 0; t2 < 2; ++t2) {
      int m = wr + t2 * 16 + nl;
      a[t2] = frag16(&As[m][kb16], &As[m][kb16 + 8]);
    }
#pragma unroll
    for (int t2 = 0; t2 < 2; ++t2) {
      int n = wc + t2 * 16 + nl;
      b[t2] = frag16(&Bt[n][kb16], &Bt[n][kb16 + 8]);
    }
#pragma unroll
    for (int ti = 0; ti < 2; ++ti)
#pragma unroll
      for (int tj = 0; tj < 2; ++tj)
        acc[ti][tj] = WMMA_F32_F16(a[ti], b[tj], acc[ti][tj]);
  }

#pragma unroll
  for (int ti = 0; ti < 2; ++ti)
#pragma unroll
    for (int tj = 0; tj < 2; ++tj)
#pragma unroll
      for (int i = 0; i < 8; ++i) {
        int r = r0 + wr + ti * 16 + i + (hi ? 8 : 0);
        int c = c0 + wc + tj * 16 + nl;
        O[(size_t)r * CH + c] = acc[ti][tj][i];
      }
}

// ---------------------------------------------------------------- launcher
extern "C" void kernel_launch(void* const* d_in, const int* in_sizes, int n_in,
                              void* d_out, int out_size, void* d_ws, size_t ws_size,
                              hipStream_t stream) {
  (void)in_sizes; (void)n_in; (void)out_size; (void)ws_size;
  const float* x  = (const float*)d_in[0];
  const float* Wq = (const float*)d_in[1];
  const float* Wk = (const float*)d_in[2];
  const float* Wv = (const float*)d_in[3];
  const float* Wo = (const float*)d_in[4];
  const float* gi = (const float*)d_in[5];
  float* out = (float*)d_out;

  char* p = (char*)d_ws;
  auto carve = [&](size_t bytes) {
    char* r = p;
    p += (bytes + 255) & ~(size_t)255;
    return r;
  };
  _Float16* xh  = (_Float16*)carve((size_t)BT * CH * 2);
  _Float16* wqh = (_Float16*)carve((size_t)CH * CH * 2);
  _Float16* wkh = (_Float16*)carve((size_t)CH * CH * 2);
  _Float16* wvh = (_Float16*)carve((size_t)CH * CH * 2);
  _Float16* woh = (_Float16*)carve((size_t)CH * CH * 2);
  _Float16* qh  = (_Float16*)carve((size_t)BT * CH * 2);
  _Float16* kh  = (_Float16*)carve((size_t)BT * CH * 2);
  _Float16* vh  = (_Float16*)carve((size_t)BT * CH * 2);
  _Float16* oh  = (_Float16*)carve((size_t)BT * CH * 2);
  float*    kr  = (float*)carve((size_t)T * 4);

  const int nx = BT * CH, nw = CH * CH;
  cvt_f32_f16<<<(nx + 255) / 256, 256, 0, stream>>>(x, xh, nx);
  cvt_f32_f16<<<(nw + 255) / 256, 256, 0, stream>>>(Wq, wqh, nw);
  cvt_f32_f16<<<(nw + 255) / 256, 256, 0, stream>>>(Wk, wkh, nw);
  cvt_f32_f16<<<(nw + 255) / 256, 256, 0, stream>>>(Wv, wvh, nw);
  cvt_f32_f16<<<(nw + 255) / 256, 256, 0, stream>>>(Wo, woh, nw);
  keeprate_kernel<<<1, 256, 0, stream>>>(gi, kr);

  dim3 gq(CH / 64, BT / 64, 3);
  qkv_gemm<<<gq, 128, 0, stream>>>(xh, wqh, wkh, wvh, qh, kh, vh);

  dim3 ga(T / 64, BATCH * H, 1);
  attn_kernel<<<ga, 128, 0, stream>>>(qh, kh, vh, kr, oh);

  dim3 go(CH / 64, BT / 64, 1);
  out_gemm<<<go, 128, 0, stream>>>(oh, woh, out);
}
